// AttentionLayer_10264971837964
// MI455X (gfx1250) — matrix operand order repaired and checked by reading, not compiled, not run on hardware
//
#include <hip/hip_runtime.h>

// CDNA5 (gfx1250) fused attention: scores = Q@K^T (fp32 out), softmax, context = P@K.
// bf16 WMMA (v_wmma_f32_16x16x32_bf16) with fp32 accumulation for both GEMMs.
// fp32->bf16 via native V_CVT_PK_BF16_F32 instead of manual bit rounding.

typedef __attribute__((ext_vector_type(16))) __bf16 v16bf;
typedef __attribute__((ext_vector_type(2)))  __bf16 v2bf;
typedef __attribute__((ext_vector_type(8)))  float  v8f;

union BF16x16 { v16bf v; unsigned int u[8]; };

static constexpr int B_  = 16;
static constexpr int TQ_ = 1024;
static constexpr int TK_ = 1024;
static constexpr int H_  = 1024;

// Two fp32 -> packed bf16 dword, using the hardware pack-convert.
__device__ __forceinline__ unsigned int cvt2(float lo, float hi) {
  union { v2bf v; unsigned int u; } r;
#if __has_builtin(__builtin_amdgcn_cvt_pk_bf16_f32)
  r.v = __builtin_amdgcn_cvt_pk_bf16_f32(lo, hi);
#else
  v2bf t; t[0] = (__bf16)lo; t[1] = (__bf16)hi;
  r.v = t;
#endif
  return r.u;
}
__device__ __forceinline__ unsigned short cvt1(float x) {
  union { __bf16 h; unsigned short s; } r;
  r.h = (__bf16)x;
  return r.s;
}
__device__ __forceinline__ float bf2f(unsigned short h) {
  union { unsigned short s; __bf16 h; } r;
  r.s = h;
  return (float)r.h;
}

// A-matrix fragment (16x32 bf16) from a 16 x H_ bf16 row-major LDS buffer.
// Layout: lanes 0-15 hold row M=lane, K = k0+{0..15};
//         lanes 16-31 hold row M=lane-16, K = k0+{16..31}.
__device__ __forceinline__ v16bf load_A_lds(const unsigned int* sQd, int lane, int k0) {
  const int row = lane & 15;
  const unsigned int* p = sQd + row * (H_ / 2) + (k0 >> 1) + ((lane >> 4) << 3);
  BF16x16 A;
  A.u[0] = p[0];  A.u[1] = p[1];  A.u[2] = p[2];  A.u[3] = p[3];
  A.u[4] = p[4];  A.u[5] = p[5];  A.u[6] = p[6];  A.u[7] = p[7];
  return A.v;
}

// B-matrix fragment lane data: 16 consecutive fp32 -> 16 packed bf16.
// Layout used here (same k order as A): lanes 0-15 hold col N=lane, K=0..15; lanes 16-31 col N=lane-16, K=16..31.
__device__ __forceinline__ v16bf load_B_global16(const float* __restrict__ p) {
  BF16x16 Bv;
#pragma unroll
  for (int i = 0; i < 4; ++i) {
    float4 f = ((const float4*)p)[i];
    Bv.u[2 * i + 0] = cvt2(f.x, f.y);
    Bv.u[2 * i + 1] = cvt2(f.z, f.w);
  }
  return Bv.v;
}

__global__ __launch_bounds__(256) void attn_fused_kernel(
    const float* __restrict__ q, const float* __restrict__ kmem,
    float* __restrict__ ctx_out, float* __restrict__ sc_out) {
  extern __shared__ char smem[];
  unsigned short* sQ  = (unsigned short*)smem;                       // 16 x 1024 bf16 (32 KB), later probs
  unsigned int*   sQd = (unsigned int*)smem;
  float*          red = (float*)(smem + 16 * H_ * 2);                // 256 f32 (1 KB)
  unsigned short* sKT = (unsigned short*)(smem + 16 * H_ * 2 + 1024);// 128 x 34 bf16 (8.5 KB), K^T staging

  const int tid  = threadIdx.x;
  const int wave = tid >> 5;
  const int lane = tid & 31;
  const int b    = blockIdx.x >> 6;           // TQ/16 = 64 q-tiles per batch
  const int q0   = (blockIdx.x & 63) << 4;

  const float* qt  = q    + ((size_t)b * TQ_ + q0) * (size_t)H_;
  const float* kb  = kmem + (size_t)b * TK_ * (size_t)H_;
  float*       sct = sc_out  + ((size_t)b * TQ_ + q0) * (size_t)TK_;
  float*       cxt = ctx_out + ((size_t)b * TQ_ + q0) * (size_t)H_;

  // ---- Stage Q tile (16 x H) into LDS as bf16 ----
  for (int i = tid; i < 16 * H_ / 4; i += 256) {
    const int e   = i << 2;
    const int row = e >> 10;
    const int c   = e & (H_ - 1);
    float4 v = *(const float4*)(qt + (size_t)row * H_ + c);
    const int o2 = (row * H_ + c) >> 1;
    sQd[o2]     = cvt2(v.x, v.y);
    sQd[o2 + 1] = cvt2(v.z, v.w);
  }
  __syncthreads();

  // ---- Pass 1: raw scores, each wave owns 8 contiguous 16-wide k tiles ----
#pragma unroll 1
  for (int j = 0; j < 8; ++j) {
    const int kt = wave * 8 + j;
    v8f acc = {};
    const float* bbase =
        kb + (size_t)(kt * 16 + (lane & 15)) * H_ + ((lane >> 4) << 4);
#pragma unroll 4
    for (int h0 = 0; h0 < H_; h0 += 32) {
      __builtin_prefetch(bbase + h0 + 128, 0, 1);   // global_prefetch_b8 ahead in K row
      v16bf A  = load_A_lds(sQd, lane, h0);
      v16bf Bm = load_B_global16(bbase + h0);
      acc = __builtin_amdgcn_wmma_f32_16x16x32_bf16(false, A, false, Bm,
                                                    (short)0, acc, false, false);
    }
    // C/D layout: lanes 0-15 -> N=lane, M=v; lanes 16-31 -> N=lane-16, M=v+8
    const int col = lane & 15;
    const int mb  = (lane >> 4) << 3;
    float* dst = sct + (size_t)mb * TK_ + kt * 16 + col;
#pragma unroll
    for (int v = 0; v < 8; ++v) dst[(size_t)v * TK_] = acc[v];
  }

  __threadfence_block();
  __syncthreads();

  // ---- Softmax over key dim (scores read back through L2); probs -> sQ as bf16 ----
  {
    const int row = tid >> 4;
    const int t   = tid & 15;
    const float* srow = sct + (size_t)row * TK_;
    float m = -3.402823466e38f;
    for (int c = t; c < TK_; c += 16) m = fmaxf(m, srow[c]);
    red[tid] = m;
    __syncthreads();
#pragma unroll
    for (int s = 8; s > 0; s >>= 1) {
      if (t < s) red[tid] = fmaxf(red[tid], red[tid + s]);
      __syncthreads();
    }
    const float rmax = red[row << 4];
    __syncthreads();
    float ssum = 0.f;
    for (int c = t; c < TK_; c += 16) {
      const float e = __expf(srow[c] - rmax);
      ssum += e;
      sQ[row * TK_ + c] = cvt1(e);
    }
    red[tid] = ssum;
    __syncthreads();
#pragma unroll
    for (int s = 8; s > 0; s >>= 1) {
      if (t < s) red[tid] += red[tid + s];
      __syncthreads();
    }
    const float inv = 1.f / red[row << 4];
    __syncthreads();
    for (int c = t; c < TK_; c += 16) {
      sQ[row * TK_ + c] = cvt1(bf2f(sQ[row * TK_ + c]) * inv);
    }
  }
  __syncthreads();

  // ---- Pass 2: context = P(16 x TK) @ K(TK x H); K chunk transposed through LDS ----
#pragma unroll 1
  for (int j = 0; j < 8; ++j) {
    const int nt = 8 * j + wave;             // this wave's 16-wide h tile
    v8f acc = {};
#pragma unroll 1
    for (int k0 = 0; k0 < TK_; k0 += 32) {
      __syncthreads();                       // previous chunk fully consumed
      // Stage K[k0..k0+31][128j .. 128j+127] transposed: sKT[col][krow], stride 34
      for (int it = tid; it < 32 * 32; it += 256) {
        const int kr = it >> 5;              // 0..31
        const int cb = (it & 31) << 2;       // 0..124, step 4
        float4 f = *(const float4*)(kb + (size_t)(k0 + kr) * H_ + (j << 7) + cb);
        sKT[(cb + 0) * 34 + kr] = cvt1(f.x);
        sKT[(cb + 1) * 34 + kr] = cvt1(f.y);
        sKT[(cb + 2) * 34 + kr] = cvt1(f.z);
        sKT[(cb + 3) * 34 + kr] = cvt1(f.w);
      }
      __syncthreads();
      v16bf A = load_A_lds(sQd, lane, k0);   // probs fragment
      // B fragment: local col = wave*16 + (lane&15), K base = (lane>>4)*16
      const unsigned int* bp = (const unsigned int*)sKT;
      const int c  = (wave << 4) + (lane & 15);
      const int d0 = c * 17 + ((lane >> 4) << 3);   // dword offset (34 ushorts/row)
      BF16x16 Bv;
#pragma unroll
      for (int i = 0; i < 8; ++i) Bv.u[i] = bp[d0 + i];
      acc = __builtin_amdgcn_wmma_f32_16x16x32_bf16(false, A, false, Bv.v,
                                                    (short)0, acc, false, false);
    }
    const int col = lane & 15;
    const int mb  = (lane >> 4) << 3;
    float* dst = cxt + (size_t)mb * H_ + nt * 16 + col;
#pragma unroll
    for (int v = 0; v < 8; ++v) dst[(size_t)v * H_] = acc[v];
  }
}

extern "C" void kernel_launch(void* const* d_in, const int* in_sizes, int n_in,
                              void* d_out, int out_size, void* d_ws, size_t ws_size,
                              hipStream_t stream) {
  (void)in_sizes; (void)n_in; (void)out_size; (void)d_ws; (void)ws_size;
  const float* q    = (const float*)d_in[0];
  const float* kmem = (const float*)d_in[1];
  float* ctx = (float*)d_out;                               // [B,TQ,H]
  float* sc  = ctx + (size_t)B_ * TQ_ * H_;                 // [B,TQ,TK]

  const dim3 grid(B_ * (TQ_ / 16));                         // 1024 blocks
  const dim3 block(256);                                    // 8 wave32
  const size_t shmem = (size_t)16 * H_ * 2 + 1024 + 128 * 34 * 2;  // ~42.5 KB
  hipLaunchKernelGGL(attn_fused_kernel, grid, block, shmem, stream, q, kmem, ctx, sc);
}
